// CCConvLayer_73959336837364
// MI455X (gfx1250) — matrix operand order repaired and checked by reading, not compiled, not run on hardware
//
#include <hip/hip_runtime.h>
#include <hip/hip_fp16.h>

typedef __attribute__((ext_vector_type(16))) _Float16 v16h;
typedef __attribute__((ext_vector_type(8)))  float    v8f;

#define IN_CH  128
#define OUT_CH 128

union Frag16 { v16h v; uint4 q[2]; };

// ---------------------------------------------------------------------------
// 1) Stage x and W as f16 so WMMA fragments are straight b128 loads.
// ---------------------------------------------------------------------------
__global__ void cc_cvt_kernel(const float* __restrict__ x, const float* __restrict__ W,
                              _Float16* __restrict__ xh, _Float16* __restrict__ Wh,
                              int nX, int nW) {
    int i = blockIdx.x * blockDim.x + threadIdx.x;
    if (i < nX) xh[i] = (_Float16)x[i];
    if (i < nW) Wh[i] = (_Float16)W[i];
}

// ---------------------------------------------------------------------------
// 2) x1 = x @ W^T via v_wmma_f32_16x16x32_f16.
//    One wave per 16-row M-tile; 8 N-tiles (128 out ch), 4 K-steps of 32.
// ---------------------------------------------------------------------------
__global__ void cc_gemm_kernel(const _Float16* __restrict__ xh,
                               const _Float16* __restrict__ Wh,
                               float* __restrict__ x1, int nNodes) {
    const int lane = threadIdx.x & 31;
    const int wave = threadIdx.x >> 5;
    const int tile = blockIdx.x * (blockDim.x >> 5) + wave;
    const int nTiles = (nNodes + 15) >> 4;
    if (tile >= nTiles) return;                 // wave-uniform: EXEC stays all-1s

    const int m  = lane & 15;
    const int hi = lane >> 4;
    const int rowBase = tile << 4;

    int rowA = rowBase + m;
    if (rowA >= nNodes) rowA = nNodes - 1;      // clamp (keeps EXEC full)
    const _Float16* xrow = xh + (size_t)rowA * IN_CH;

    v8f acc[8];
#pragma unroll
    for (int t = 0; t < 8; ++t) acc[t] = (v8f){0.f,0.f,0.f,0.f,0.f,0.f,0.f,0.f};

#pragma unroll
    for (int kt = 0; kt < 4; ++kt) {
        const int kbase = kt * 32;
        // A-frag: 16x32 f16, halfs 0-7 = K[kbase+8*hi ..], halfs 8-15 = K[kbase+16+8*hi ..]
        Frag16 a;
        a.q[0] = *(const uint4*)(xrow + kbase + 8 * hi);
        a.q[1] = *(const uint4*)(xrow + kbase + 16 + 8 * hi);
#pragma unroll
        for (int nt = 0; nt < 8; ++nt) {
            // B-frag: 32x16 f16, lane n=m holds W[16*nt+m][kbase+8*hi ..+7] and [kbase+16+8*hi ..+7]
            const _Float16* wrow = Wh + (size_t)(nt * 16 + m) * IN_CH + kbase;
            Frag16 b;
            b.q[0] = *(const uint4*)(wrow + 8 * hi);
            b.q[1] = *(const uint4*)(wrow + 16 + 8 * hi);
            acc[nt] = __builtin_amdgcn_wmma_f32_16x16x32_f16(
                false, a.v, false, b.v, (short)0, acc[nt], false, false);
        }
    }

    // D layout: VGPR r -> row (rowBase + r + 8*hi), col (nt*16 + m)
#pragma unroll
    for (int nt = 0; nt < 8; ++nt) {
#pragma unroll
        for (int r = 0; r < 8; ++r) {
            int row = rowBase + r + 8 * hi;
            if (row < nNodes)
                x1[(size_t)row * OUT_CH + nt * 16 + m] = acc[nt][r];
        }
    }
}

// ---------------------------------------------------------------------------
// 3) SpMM scatter: one wave per edge, 4 channels/lane, HW f32 atomics.
//    x1 (20.5 MB) and out (20.5 MB) are L2-resident on MI455X (192 MB L2).
// ---------------------------------------------------------------------------
__global__ void cc_scatter_kernel(const float* __restrict__ x1,
                                  const int* __restrict__ src,
                                  const int* __restrict__ dst,
                                  const float* __restrict__ vals,
                                  float* __restrict__ out, int nE) {
    int gid  = blockIdx.x * blockDim.x + threadIdx.x;
    int e    = gid >> 5;
    int lane = gid & 31;
    if (e >= nE) return;
    int   s = src[e];
    int   d = dst[e];
    float v = vals[e];
    float4 g = ((const float4*)(x1 + (size_t)s * OUT_CH))[lane];
    float* o = out + (size_t)d * OUT_CH + lane * 4;
    unsafeAtomicAdd(o + 0, g.x * v);
    unsafeAtomicAdd(o + 1, g.y * v);
    unsafeAtomicAdd(o + 2, g.z * v);
    unsafeAtomicAdd(o + 3, g.w * v);
}

// ---------------------------------------------------------------------------
// 4) ReLU in place.
// ---------------------------------------------------------------------------
__global__ void cc_relu_kernel(float* __restrict__ out, int n) {
    int i = blockIdx.x * blockDim.x + threadIdx.x;
    if (i < n) out[i] = fmaxf(out[i], 0.0f);
}

extern "C" void kernel_launch(void* const* d_in, const int* in_sizes, int n_in,
                              void* d_out, int out_size, void* d_ws, size_t ws_size,
                              hipStream_t stream) {
    const float* x    = (const float*)d_in[0];
    const float* W    = (const float*)d_in[1];
    const int*   src  = (const int*)d_in[2];
    const int*   dst  = (const int*)d_in[3];
    const float* vals = (const float*)d_in[4];

    const int nX = in_sizes[0];            // nNodes * IN_CH
    const int nW = in_sizes[1];            // OUT_CH * IN_CH
    const int nE = in_sizes[2];
    const int nNodes = nX / IN_CH;
    float* out = (float*)d_out;

    // Workspace layout: x1 (f32) | xh (f16) | Wh (f16), 256B-aligned sections.
    char* ws = (char*)d_ws;
    size_t off1 = ((size_t)nX * 4 + 255) & ~(size_t)255;
    size_t off2 = (off1 + (size_t)nX * 2 + 255) & ~(size_t)255;
    float*     x1 = (float*)ws;
    _Float16*  xh = (_Float16*)(ws + off1);
    _Float16*  Wh = (_Float16*)(ws + off2);
    (void)ws_size;

    // 1) convert to f16
    int nMax = nX > nW ? nX : nW;
    cc_cvt_kernel<<<(nMax + 255) / 256, 256, 0, stream>>>(x, W, xh, Wh, nX, nW);

    // 2) projection GEMM (8 waves / block)
    int nTiles = (nNodes + 15) / 16;
    cc_gemm_kernel<<<(nTiles + 7) / 8, 256, 0, stream>>>(xh, Wh, x1, nNodes);

    // 3) zero output accumulator
    hipMemsetAsync(d_out, 0, (size_t)out_size * sizeof(float), stream);

    // 4) edge scatter (one wave per edge)
    long long scatterThreads = (long long)nE * 32;
    cc_scatter_kernel<<<(unsigned)((scatterThreads + 255) / 256), 256, 0, stream>>>(
        x1, src, dst, vals, out, nE);

    // 5) ReLU
    cc_relu_kernel<<<(out_size + 255) / 256, 256, 0, stream>>>(out, out_size);
}
